// LearnGeLU_26508538151453
// MI455X (gfx1250) — compile-verified, matrix operand order checked by reading, not run on hardware
//
#include <hip/hip_runtime.h>
#include <hip/hip_bf16.h>

typedef __bf16 bf16_t;
typedef __bf16 v16bf __attribute__((ext_vector_type(16)));
typedef __bf16 v8bf  __attribute__((ext_vector_type(8)));
typedef __bf16 v4bf  __attribute__((ext_vector_type(4)));
typedef float  v8f   __attribute__((ext_vector_type(8)));

#define BM 128
#define BN 128
#define BK 32
#define LDT 40  // padded LDS row stride (bf16 elems): 80B -> 20-bank skew, conflict-free

// ---- CDNA5 async global->LDS copy (ASYNCcnt path), with safe fallback ------
#if defined(__AMDGCN__) && __has_builtin(__builtin_amdgcn_global_load_async_to_lds_b128)
#define ASYNC_LDS 1
#else
#define ASYNC_LDS 0
#endif

#if defined(__AMDGCN__) && __has_builtin(__builtin_amdgcn_s_wait_asynccnt)
#define WAIT_ASYNC() __builtin_amdgcn_s_wait_asynccnt(0)
#elif ASYNC_LDS
#define WAIT_ASYNC() asm volatile("s_wait_asynccnt 0x0" ::: "memory")
#else
#define WAIT_ASYNC() ((void)0)
#endif

#if ASYNC_LDS
// Builtin signature (from hipcc sema): (int4 AS1*, int4 AS3*, imm offset, imm cpol)
typedef int v4i_vs __attribute__((vector_size(16)));
#define AS1V4(p) ((__attribute__((address_space(1))) v4i_vs*)(p))
#define AS3V4(p) ((__attribute__((address_space(3))) v4i_vs*)(p))
#endif

// ---------------------------------------------------------------------------
// f32 -> bf16 conversion (vectorized x4), grid-stride
// ---------------------------------------------------------------------------
__global__ __launch_bounds__(256)
void cvt_f32_bf16(const float* __restrict__ in, bf16_t* __restrict__ out, long long n4) {
    long long i = (long long)blockIdx.x * blockDim.x + threadIdx.x;
    long long stride = (long long)gridDim.x * blockDim.x;
    for (; i < n4; i += stride) {
        float4 v = ((const float4*)in)[i];
        v4bf r;
        r[0] = (bf16_t)v.x; r[1] = (bf16_t)v.y; r[2] = (bf16_t)v.z; r[3] = (bf16_t)v.w;
        ((v4bf*)out)[i] = r;
    }
}

// ---------------------------------------------------------------------------
// bf16 WMMA GEMM:  C[M,N] = A[M,Kred] * W[N,Kred]^T  (+bias, optional gate)
// Workgroup: 256 threads (8 waves), tile 128x128, K-step 32, double-buffered
// LDS fed by GLOBAL_LOAD_ASYNC_TO_LDS_B128 (ASYNCcnt), WMMA 16x16x32 bf16.
// ---------------------------------------------------------------------------
template<bool GATED>
__global__ __launch_bounds__(256)
void wmma_gemm(const bf16_t* __restrict__ A,     // [M, Kred] row-major
               const bf16_t* __restrict__ W,     // [N, Kred] row-major
               const float*  __restrict__ bias,  // [N]
               const float*  __restrict__ scale, // [N] (GATED only)
               bf16_t* __restrict__ outH,        // [M, N] bf16 (if !GATED)
               float*  __restrict__ outF,        // [M, N] f32  (if GATED)
               int M, int N, int Kred)
{
    __shared__ bf16_t lA[2][BM * LDT];
    __shared__ bf16_t lB[2][BN * LDT];

    const int tid  = threadIdx.x;
    const int lane = tid & 31;
    const int wave = tid >> 5;
    const int wr   = wave & 3;        // wave row group  -> M offset wr*32
    const int wc   = wave >> 2;       // wave col group  -> N offset wc*64
    const int m0   = blockIdx.y * BM;
    const int n0   = blockIdx.x * BN;

    const int half  = lane >> 4;      // 0: lanes 0-15, 1: lanes 16-31
    const int l16   = lane & 15;
    const int khalf = half * 8;       // K sub-offset per ISA 16-bit A/B layout

    v8f acc[2][4];
    #pragma unroll
    for (int mi = 0; mi < 2; ++mi)
        #pragma unroll
        for (int ni = 0; ni < 4; ++ni)
            acc[mi][ni] = (v8f){0.f,0.f,0.f,0.f,0.f,0.f,0.f,0.f};

    const int c0 = tid * 2;           // each thread moves 2x16B chunks per tile per matrix
    const int nk = Kred / BK;

    auto loadTile = [&](int buf, int kt) {
        const int kbase = kt * BK;
        #pragma unroll
        for (int u = 0; u < 2; ++u) {
            const int c   = c0 + u;
            const int row = c >> 2;
            const int col = (c & 3) * 8;
#if ASYNC_LDS
            __builtin_amdgcn_global_load_async_to_lds_b128(
                AS1V4(A + (size_t)(m0 + row) * Kred + kbase + col),
                AS3V4(&lA[buf][row * LDT + col]),
                0, 0);
            __builtin_amdgcn_global_load_async_to_lds_b128(
                AS1V4(W + (size_t)(n0 + row) * Kred + kbase + col),
                AS3V4(&lB[buf][row * LDT + col]),
                0, 0);
#else
            uint4 va = *(const uint4*)(A + (size_t)(m0 + row) * Kred + kbase + col);
            *(uint4*)(&lA[buf][row * LDT + col]) = va;
            uint4 vb = *(const uint4*)(W + (size_t)(n0 + row) * Kred + kbase + col);
            *(uint4*)(&lB[buf][row * LDT + col]) = vb;
#endif
        }
    };

    auto computeTile = [&](int buf) {
        v16bf afrag[2];
        v16bf bfrag[4];
        #pragma unroll
        for (int mi = 0; mi < 2; ++mi) {
            const bf16_t* p = &lA[buf][(wr * 32 + mi * 16 + l16) * LDT + khalf];
            v8bf lo = *(const v8bf*)(p);        // K = khalf .. khalf+7
            v8bf hi = *(const v8bf*)(p + 16);   // K = khalf+16 .. khalf+23
            afrag[mi] = __builtin_shufflevector(lo, hi, 0,1,2,3,4,5,6,7,8,9,10,11,12,13,14,15);
        }
        #pragma unroll
        for (int ni = 0; ni < 4; ++ni) {
            const bf16_t* p = &lB[buf][(wc * 64 + ni * 16 + l16) * LDT + khalf];
            v8bf lo = *(const v8bf*)(p);
            v8bf hi = *(const v8bf*)(p + 16);
            bfrag[ni] = __builtin_shufflevector(lo, hi, 0,1,2,3,4,5,6,7,8,9,10,11,12,13,14,15);
        }
        #pragma unroll
        for (int mi = 0; mi < 2; ++mi)
            #pragma unroll
            for (int ni = 0; ni < 4; ++ni)
                acc[mi][ni] = __builtin_amdgcn_wmma_f32_16x16x32_bf16(
                    false, afrag[mi], false, bfrag[ni],
                    (short)0, acc[mi][ni], false, false);
    };

    loadTile(0, 0);
    int buf = 0;
    for (int kt = 0; kt < nk; ++kt) {
        // Make tile[buf] visible to all waves: drain my async copies, then barrier.
        WAIT_ASYNC();
        __syncthreads();
        if (kt + 1 < nk) loadTile(buf ^ 1, kt + 1);
        if (kt + 2 < nk) {  // global_prefetch_b8 of the K+2 tile
            const int row = c0 >> 2, col = (c0 & 3) * 8;
            __builtin_prefetch(A + (size_t)(m0 + row) * Kred + (kt + 2) * BK + col, 0, 1);
            __builtin_prefetch(W + (size_t)(n0 + row) * Kred + (kt + 2) * BK + col, 0, 1);
        }
        computeTile(buf);
        buf ^= 1;
    }

    // Epilogue. D layout (32-bit C/D 16x16): lane<16 -> N=l16, M=r; lane>=16 -> M=r+8.
    #pragma unroll
    for (int mi = 0; mi < 2; ++mi) {
        #pragma unroll
        for (int ni = 0; ni < 4; ++ni) {
            const int n = n0 + wc * 64 + ni * 16 + l16;
            const float bv = bias[n];
            float sv = 0.f;
            if (GATED) sv = scale[n];
            #pragma unroll
            for (int r = 0; r < 8; ++r) {
                const int m = m0 + wr * 32 + mi * 16 + r + (half ? 8 : 0);
                const float v = acc[mi][ni][r] + bv;
                if (GATED) {
                    outF[(size_t)m * N + n] = sv / (1.0f + __expf(-v));
                } else {
                    outH[(size_t)m * N + n] = (bf16_t)v;
                }
            }
        }
    }
}

// ---------------------------------------------------------------------------
// Exact per-row top-k threshold via 4-pass radix select on positive-f32 bit
// patterns (monotone as u32), then in-place mask of d_out. 1 workgroup / row.
// ---------------------------------------------------------------------------
__global__ __launch_bounds__(256)
void topk_mask(float* __restrict__ out, int rowLen, int k) {
    const int row = blockIdx.x;
    const unsigned* keys = (const unsigned*)(out + (size_t)row * rowLen);

    __shared__ unsigned hist[256];
    __shared__ unsigned s_bin, s_krem;

    unsigned prefix = 0u, hiMask = 0u;
    unsigned krem = (unsigned)k;

    for (int pass = 3; pass >= 0; --pass) {
        const int shift = pass * 8;
        for (int i = threadIdx.x; i < 256; i += blockDim.x) hist[i] = 0u;
        __syncthreads();
        for (int i = threadIdx.x; i < rowLen; i += blockDim.x) {
            const unsigned key = keys[i];
            if ((key & hiMask) == prefix)
                atomicAdd(&hist[(key >> shift) & 255u], 1u);
        }
        __syncthreads();
        if (threadIdx.x == 0) {
            unsigned cum = 0u; int b = 255;
            for (; b > 0; --b) {
                const unsigned h = hist[b];
                if (cum + h >= krem) break;
                cum += h;
            }
            s_bin  = (unsigned)b;
            s_krem = krem - cum;
        }
        __syncthreads();
        prefix |= s_bin << shift;
        hiMask |= 0xFFu << shift;
        krem = s_krem;
        __syncthreads();
    }

    const unsigned thr = prefix;  // exact k-th largest bit pattern
    for (int i = threadIdx.x; i < rowLen; i += blockDim.x) {
        const unsigned key = keys[i];
        out[(size_t)row * rowLen + i] = (key >= thr) ? __uint_as_float(key) : 0.0f;
    }
}

// ---------------------------------------------------------------------------
// Host launcher
// ---------------------------------------------------------------------------
extern "C" void kernel_launch(void* const* d_in, const int* in_sizes, int n_in,
                              void* d_out, int out_size, void* d_ws, size_t ws_size,
                              hipStream_t stream) {
    const float* x     = (const float*)d_in[0];   // [8192, 4096]
    const float* W1    = (const float*)d_in[1];   // [1024, 4096]
    const float* b1    = (const float*)d_in[2];   // [1024]
    const float* W2    = (const float*)d_in[3];   // [16384, 1024]
    const float* b2    = (const float*)d_in[4];   // [16384]
    const float* scale = (const float*)d_in[5];   // [16384]

    const int Bsz = 8192, IN = 4096, K = 1024, OUT = 16384, TOPK = 1638;

    char* ws = (char*)d_ws;
    bf16_t* xb  = (bf16_t*)ws;  ws += (size_t)Bsz * IN  * sizeof(bf16_t);
    bf16_t* w1b = (bf16_t*)ws;  ws += (size_t)K   * IN  * sizeof(bf16_t);
    bf16_t* w2b = (bf16_t*)ws;  ws += (size_t)OUT * K   * sizeof(bf16_t);
    bf16_t* hb  = (bf16_t*)ws;  // [8192, 1024] bf16 hidden

    cvt_f32_bf16<<<4096, 256, 0, stream>>>(x,  xb,  (long long)Bsz * IN  / 4);
    cvt_f32_bf16<<<2048, 256, 0, stream>>>(W1, w1b, (long long)K   * IN  / 4);
    cvt_f32_bf16<<<4096, 256, 0, stream>>>(W2, w2b, (long long)OUT * K   / 4);

    // h = x @ W1^T + b1   -> bf16
    wmma_gemm<false><<<dim3(K / BN, Bsz / BM), 256, 0, stream>>>(
        xb, w1b, b1, nullptr, hb, nullptr, Bsz, K, IN);

    // gated = sigmoid(h @ W2^T + b2) * scale -> f32 into d_out
    wmma_gemm<true><<<dim3(OUT / BN, Bsz / BM), 256, 0, stream>>>(
        hb, w2b, b2, scale, nullptr, (float*)d_out, Bsz, OUT, K);

    // keep top-1638 per row, zero the rest (in place)
    topk_mask<<<Bsz, 256, 0, stream>>>((float*)d_out, OUT, TOPK);
}
